// QuantizedLinear_8512625180852
// MI455X (gfx1250) — compile-verified, matrix operand order checked by reading, not run on hardware
//
#include <hip/hip_runtime.h>
#include <hip/hip_bf16.h>
#include <stdint.h>

typedef __attribute__((ext_vector_type(16))) __bf16 v16bf;
typedef __attribute__((ext_vector_type(8)))  float  v8f;
typedef __attribute__((ext_vector_type(4)))  int    v4i;
typedef unsigned long long ull;

struct __attribute__((aligned(16))) ULL2 { ull x, y; };

#define NIN   4096
#define NOUT  4096
#define ROWS  8192
#define RANK  64

#if __has_builtin(__builtin_amdgcn_global_load_async_to_lds_b128)
#define USE_ASYNC 1
#warning "CDNA5: using global_load_async_to_lds_b128 tile feed (triple-buffered)"
#else
#define USE_ASYNC 0
#warning "CDNA5: async-to-LDS builtin NOT available; using sync staging path"
#endif

__device__ __forceinline__ unsigned short f2bf(float f) {
  unsigned u = __float_as_uint(f);
  u += 0x7FFFu + ((u >> 16) & 1u);          // round-to-nearest-even
  return (unsigned short)(u >> 16);
}

// ---------------------------------------------------------------------------
// Kernel 1: x' = fwht(input / scaleWH * SU) * (1/sqrt(4096)) -> bf16
// ---------------------------------------------------------------------------
__global__ __launch_bounds__(256) void k_pre_fwht(
    const float* __restrict__ in, const float* __restrict__ scaleWH,
    const float* __restrict__ SU, unsigned short* __restrict__ Xb) {
  __shared__ float buf[NIN];
  const int row = blockIdx.x, tid = threadIdx.x;
  const float* src = in + (size_t)row * NIN;
  const int c0 = tid * 16;
  #pragma unroll
  for (int e = 0; e < 16; ++e) {
    int c = c0 + e;
    buf[c] = src[c] / scaleWH[c] * SU[c];
  }
  for (int s = 0; s < 12; ++s) {
    __syncthreads();
    const int h = 1 << s;
    #pragma unroll
    for (int t = tid; t < 2048; t += 256) {
      int i = ((t >> s) << (s + 1)) | (t & (h - 1));
      float a = buf[i], b = buf[i + h];
      buf[i] = a + b; buf[i + h] = a - b;
    }
  }
  __syncthreads();
  ull* dst = (ull*)(Xb + (size_t)row * NIN + c0);
  #pragma unroll
  for (int q = 0; q < 4; ++q) {
    float f0 = buf[c0 + q * 4 + 0] * 0.015625f;
    float f1 = buf[c0 + q * 4 + 1] * 0.015625f;
    float f2 = buf[c0 + q * 4 + 2] * 0.015625f;
    float f3 = buf[c0 + q * 4 + 3] * 0.015625f;
    dst[q] = (ull)f2bf(f0) | ((ull)f2bf(f1) << 16) |
             ((ull)f2bf(f2) << 32) | ((ull)f2bf(f3) << 48);
  }
}

// ---------------------------------------------------------------------------
// Kernel 2: W_eff[i,j] = Wscale * D4_CB[Qidxs[i, j/4]][j%4] + (A@B)[i,j] -> bf16
// ---------------------------------------------------------------------------
__global__ __launch_bounds__(256) void k_weff(
    const int* __restrict__ Qidxs, const float* __restrict__ D4_CB,
    const float* __restrict__ A, const float* __restrict__ Bm,
    const float* __restrict__ WscaleP, unsigned short* __restrict__ Wb) {
  __shared__ float Af[RANK];
  const int i = blockIdx.x >> 2;
  const int g = ((blockIdx.x & 3) << 8) + threadIdx.x;   // group 0..1023
  if (threadIdx.x < RANK) Af[threadIdx.x] = A[(size_t)i * RANK + threadIdx.x];
  __syncthreads();
  const float ws = WscaleP[0];
  const int   q  = Qidxs[(size_t)i * (NIN / 4) + g];
  const int   j0 = g * 4;
  float lr0 = 0.f, lr1 = 0.f, lr2 = 0.f, lr3 = 0.f;
  #pragma unroll 8
  for (int k = 0; k < RANK; ++k) {
    const float ak = Af[k];
    const float4 bv = *(const float4*)(Bm + (size_t)k * NIN + j0);
    lr0 += ak * bv.x; lr1 += ak * bv.y; lr2 += ak * bv.z; lr3 += ak * bv.w;
  }
  const float4 cb = *(const float4*)(D4_CB + (size_t)q * 4);
  const float w0 = ws * cb.x + lr0, w1 = ws * cb.y + lr1;
  const float w2 = ws * cb.z + lr2, w3 = ws * cb.w + lr3;
  *(ull*)(Wb + (size_t)i * NIN + j0) =
      (ull)f2bf(w0) | ((ull)f2bf(w1) << 16) |
      ((ull)f2bf(w2) << 32) | ((ull)f2bf(w3) << 48);
}

// ---------------------------------------------------------------------------
// Kernel 3: z = x' @ W_eff^T   (M=8192, N=4096, K=4096, bf16 in, f32 out)
// 256 threads = 8 wave32; WG tile 128x128, K-step 32.
// Triple-buffered LDS fed by GLOBAL_LOAD_ASYNC_TO_LDS_B128; counted
// s_wait_asynccnt 4 keeps one tile in flight across each compute phase.
// ---------------------------------------------------------------------------
#define BM 128
#define BN 128
#define BK 32
#define LDT 40   // LDS row stride in bf16: 80 B = 16B-aligned, conflict-free spread
#define NBUF 3

union Frag { v16bf v; ull u[4]; };

#if USE_ASYNC
typedef __attribute__((address_space(1))) v4i* gv4i_p;
typedef __attribute__((address_space(3))) v4i* lv4i_p;
__device__ __forceinline__ void async_b128(const unsigned short* g, unsigned short* l) {
  __builtin_amdgcn_global_load_async_to_lds_b128((gv4i_p)g, (lv4i_p)l, 0, 0);
}
// wait until at most the newest 4 async loads remain outstanding
__device__ __forceinline__ void wait_async_le4() {
  asm volatile("s_wait_asynccnt 0x4" ::: "memory");
}
__device__ __forceinline__ void wait_async0() {
  asm volatile("s_wait_asynccnt 0x0" ::: "memory");
}
#endif

__global__ __launch_bounds__(256) void k_gemm(
    const unsigned short* __restrict__ Xb,
    const unsigned short* __restrict__ Wb, float* __restrict__ C) {
  __shared__ unsigned short sX[NBUF][BM * LDT];
  __shared__ unsigned short sW[NBUF][BN * LDT];

  const int tid  = threadIdx.x;
  const int lane = tid & 31, wave = tid >> 5;
  const int half = lane >> 4, r = lane & 15;     // WMMA lane mapping
  const int wm = wave >> 1, wn = wave & 1;       // 4x2 wave grid
  const int bM = blockIdx.y * BM, bN = blockIdx.x * BN;

  const int ldRow = tid >> 2;                    // 0..63 (two passes of 64 rows)
  const int ldCol = (tid & 3) * 8;               // bf16 column within K-step
  const size_t xbase = (size_t)(bM + ldRow) * NIN + ldCol;
  const size_t wbase = (size_t)(bN + ldRow) * NIN + ldCol;
  const int lofs0 = ldRow * LDT + ldCol;
  const int lofs1 = (ldRow + 64) * LDT + ldCol;

  const int NK = NIN / BK;

#if USE_ASYNC
  // prologue: tiles 0 and 1 in flight; wait for tile 0 only (oldest 4)
  async_b128(Xb + xbase,                    &sX[0][lofs0]);
  async_b128(Xb + xbase + (size_t)64 * NIN, &sX[0][lofs1]);
  async_b128(Wb + wbase,                    &sW[0][lofs0]);
  async_b128(Wb + wbase + (size_t)64 * NIN, &sW[0][lofs1]);
  async_b128(Xb + xbase + BK,                    &sX[1][lofs0]);
  async_b128(Xb + xbase + (size_t)64 * NIN + BK, &sX[1][lofs1]);
  async_b128(Wb + wbase + BK,                    &sW[1][lofs0]);
  async_b128(Wb + wbase + (size_t)64 * NIN + BK, &sW[1][lofs1]);
  wait_async_le4();
  __syncthreads();
#else
  ULL2 rx0, rx1, rw0, rw1;
  rx0 = *(const ULL2*)(Xb + xbase);
  rx1 = *(const ULL2*)(Xb + xbase + (size_t)64 * NIN);
  rw0 = *(const ULL2*)(Wb + wbase);
  rw1 = *(const ULL2*)(Wb + wbase + (size_t)64 * NIN);
  {
    ull* p0 = (ull*)&sX[0][lofs0]; p0[0] = rx0.x; p0[1] = rx0.y;
    ull* p1 = (ull*)&sX[0][lofs1]; p1[0] = rx1.x; p1[1] = rx1.y;
    ull* q0 = (ull*)&sW[0][lofs0]; q0[0] = rw0.x; q0[1] = rw0.y;
    ull* q1 = (ull*)&sW[0][lofs1]; q1[0] = rw1.x; q1[1] = rw1.y;
  }
  __syncthreads();
#endif

  v8f acc[2][4] = {};
  int cur = 0;                    // buffer holding tile ks
  for (int ks = 0; ks < NK; ++ks) {
#if USE_ASYNC
    if (ks + 2 < NK) {
      // issue tile ks+2 into the buffer freed by iteration ks-1
      const int b2 = (cur + 2 >= NBUF) ? cur + 2 - NBUF : cur + 2;
      const size_t off = (size_t)(ks + 2) * BK;
      async_b128(Xb + xbase + off,                    &sX[b2][lofs0]);
      async_b128(Xb + xbase + (size_t)64 * NIN + off, &sX[b2][lofs1]);
      async_b128(Wb + wbase + off,                    &sW[b2][lofs0]);
      async_b128(Wb + wbase + (size_t)64 * NIN + off, &sW[b2][lofs1]);
      if (ks + 4 < NK) {
        __builtin_prefetch(Xb + xbase + off + 2 * BK, 0, 0);
        __builtin_prefetch(Wb + wbase + off + 2 * BK, 0, 0);
      }
    }
#else
    const int nxt = cur ^ 1;
    if (ks + 1 < NK) {
      const size_t off = (size_t)(ks + 1) * BK;
      rx0 = *(const ULL2*)(Xb + xbase + off);
      rx1 = *(const ULL2*)(Xb + xbase + (size_t)64 * NIN + off);
      rw0 = *(const ULL2*)(Wb + wbase + off);
      rw1 = *(const ULL2*)(Wb + wbase + (size_t)64 * NIN + off);
      if (ks + 2 < NK) {
        __builtin_prefetch(Xb + xbase + off + BK, 0, 0);
        __builtin_prefetch(Wb + wbase + off + BK, 0, 0);
      }
    }
#endif

    Frag af[2], bf[4];
    #pragma unroll
    for (int mt = 0; mt < 2; ++mt) {
      const unsigned short* p = &sX[cur][(wm * 32 + mt * 16 + r) * LDT + half * 8];
      af[mt].u[0] = *(const ull*)(p);        // K 0..3  (+half*8)
      af[mt].u[1] = *(const ull*)(p + 4);    // K 4..7  (+half*8)
      af[mt].u[2] = *(const ull*)(p + 16);   // K 16..19(+half*8)
      af[mt].u[3] = *(const ull*)(p + 20);   // K 20..23(+half*8)
    }
    #pragma unroll
    for (int nt = 0; nt < 4; ++nt) {
      const unsigned short* p = &sW[cur][(wn * 64 + nt * 16 + r) * LDT + half * 8];
      bf[nt].u[0] = *(const ull*)(p);
      bf[nt].u[1] = *(const ull*)(p + 4);
      bf[nt].u[2] = *(const ull*)(p + 16);
      bf[nt].u[3] = *(const ull*)(p + 20);
    }
    #pragma unroll
    for (int mt = 0; mt < 2; ++mt)
      #pragma unroll
      for (int nt = 0; nt < 4; ++nt)
        acc[mt][nt] = __builtin_amdgcn_wmma_f32_16x16x32_bf16(
            false, af[mt].v, false, bf[nt].v, (short)0, acc[mt][nt], false, false);

#if USE_ASYNC
    if (ks + 1 < NK) {
      if (ks + 2 < NK) wait_async_le4();   // tile ks+1 landed; ks+2 stays in flight
      else             wait_async0();      // pipeline drain (last 2 iterations)
      __syncthreads();
    }
    cur = (cur + 1 >= NBUF) ? 0 : cur + 1;
#else
    if (ks + 1 < NK) {
      ull* p0 = (ull*)&sX[nxt][lofs0]; p0[0] = rx0.x; p0[1] = rx0.y;
      ull* p1 = (ull*)&sX[nxt][lofs1]; p1[0] = rx1.x; p1[1] = rx1.y;
      ull* q0 = (ull*)&sW[nxt][lofs0]; q0[0] = rw0.x; q0[1] = rw0.y;
      ull* q1 = (ull*)&sW[nxt][lofs1]; q1[0] = rw1.x; q1[1] = rw1.y;
      __syncthreads();
    }
    cur = nxt;
#endif
  }

  // C/D layout: lane -> N = lane&15, VGPR g -> M = (lane>>4)*8 + g
  #pragma unroll
  for (int mt = 0; mt < 2; ++mt) {
    #pragma unroll
    for (int nt = 0; nt < 4; ++nt) {
      const int col  = bN + wn * 64 + nt * 16 + r;
      const int row0 = bM + wm * 32 + mt * 16 + half * 8;
      float* cp = C + (size_t)row0 * NOUT + col;
      #pragma unroll
      for (int g = 0; g < 8; ++g)
        cp[(size_t)g * NOUT] = acc[mt][nt][g];
    }
  }
}

// ---------------------------------------------------------------------------
// Kernel 4: z = fwht(z) * (1/64) * SV, in place on d_out. One WG per row.
// ---------------------------------------------------------------------------
__global__ __launch_bounds__(256) void k_post_fwht(
    float* __restrict__ z, const float* __restrict__ SV) {
  __shared__ float buf[NOUT];
  const int row = blockIdx.x, tid = threadIdx.x;
  float* src = z + (size_t)row * NOUT;
  const int c0 = tid * 16;
  #pragma unroll
  for (int e = 0; e < 16; ++e) buf[c0 + e] = src[c0 + e];
  for (int s = 0; s < 12; ++s) {
    __syncthreads();
    const int h = 1 << s;
    #pragma unroll
    for (int t = tid; t < 2048; t += 256) {
      int i = ((t >> s) << (s + 1)) | (t & (h - 1));
      float a = buf[i], b = buf[i + h];
      buf[i] = a + b; buf[i + h] = a - b;
    }
  }
  __syncthreads();
  #pragma unroll
  for (int e = 0; e < 16; ++e) {
    int c = c0 + e;
    src[c] = buf[c] * 0.015625f * SV[c];
  }
}

// ---------------------------------------------------------------------------
extern "C" void kernel_launch(void* const* d_in, const int* in_sizes, int n_in_args,
                              void* d_out, int out_size, void* d_ws, size_t ws_size,
                              hipStream_t stream) {
  (void)in_sizes; (void)n_in_args; (void)out_size; (void)ws_size;
  const float* input   = (const float*)d_in[0];
  const int*   Qidxs   = (const int*)  d_in[1];
  const float* D4_CB   = (const float*)d_in[2];
  const float* SU      = (const float*)d_in[3];
  const float* SV      = (const float*)d_in[4];
  const float* Wscale  = (const float*)d_in[5];
  const float* A       = (const float*)d_in[6];
  const float* Bm      = (const float*)d_in[7];
  const float* scaleWH = (const float*)d_in[8];
  float* out = (float*)d_out;

  // Workspace: x' bf16 (8192x4096 = 64 MB) then W_eff bf16 (4096x4096 = 32 MB)
  unsigned short* Xb = (unsigned short*)d_ws;
  unsigned short* Wb = Xb + (size_t)ROWS * NIN;

  k_pre_fwht<<<ROWS, 256, 0, stream>>>(input, scaleWH, SU, Xb);
  k_weff<<<NOUT * 4, 256, 0, stream>>>(Qidxs, D4_CB, A, Bm, Wscale, Wb);
  dim3 gg(NOUT / BN, ROWS / BM);
  k_gemm<<<gg, 256, 0, stream>>>(Xb, Wb, out);
  k_post_fwht<<<ROWS, 256, 0, stream>>>(out, SV);
}
